// MultiHeadAttention_46505905881811
// MI455X (gfx1250) — compile-verified, matrix operand order checked by reading, not run on hardware
//
#include <hip/hip_runtime.h>
#include <hip/hip_bf16.h>

// ---------------------------------------------------------------------------
// MHA forward for MI455X (gfx1250, wave32, WMMA).
//   out  = (ctx @ Wo^T + bo)                    -> d_out[0 : B*NQ*DM)
//   attn = softmax(mask( (q kT)/sqrt(dk) ))     -> d_out[B*NQ*DM : +B*H*NQ*NKV)
// GEMMs: C = A(MxK,row) @ B(NxK,row)^T, bf16 WMMA 16x16x32, f32 accum,
// fp32->bf16 via V_FMA_MIXLO/HI_BF16, double-buffered LDS pipeline.
// QK^T + mask + softmax + (probs @ V) fused into ONE kernel: attn is written
// to HBM exactly once and never re-read; ctx computed from in-LDS bf16 probs.
// ---------------------------------------------------------------------------

#define DM   1024
#define H_   16
#define DK   64
#define B_   2
#define NQ_  2048
#define NKV_ 2048

typedef __bf16 v16bf __attribute__((ext_vector_type(16)));
typedef float  v8f   __attribute__((ext_vector_type(8)));

// Pack two fp32 -> two bf16 in one dword using CDNA5 mixed-precision FMA
// converts (VOP3P V_FMA_MIXLO_BF16 / V_FMA_MIXHI_BF16): d = cvt_bf16(x*1.0+0).
__device__ __forceinline__ unsigned pack_bf16(float x, float y) {
  unsigned d;
  asm("v_fma_mixlo_bf16 %0, %1, 1.0, 0\n\t"
      "v_fma_mixhi_bf16 %0, %2, 1.0, 0"
      : "=v"(d) : "v"(x), "v"(y));
  return d;
}

__device__ __forceinline__ unsigned short bf16_one(float x) {
  unsigned d;
  asm("v_fma_mixlo_bf16 %0, %1, 1.0, 0" : "=v"(d) : "v"(x));
  return (unsigned short)d;
}

#define LDSW 40   // 32 bf16 + 8 pad (80 B row stride = 5 * 16 B: aligned + conflict-free)

// ---- staging: ROWS x 32 fp32 tile; load phase and convert/store phase split
template<int ROWS> struct TileRegs { float4 v[(ROWS * 8) / 256]; };

template<int ROWS>
__device__ __forceinline__ void tile_load(TileRegs<ROWS>& r, const float* __restrict__ g,
                                          long long ld, int tid) {
#pragma unroll
  for (int it = 0; it < (ROWS * 8) / 256; ++it) {
    const int id = tid + it * 256;
    r.v[it] = *reinterpret_cast<const float4*>(g + (long long)(id >> 3) * ld + ((id & 7) << 2));
  }
}

template<int ROWS>
__device__ __forceinline__ void tile_store(const TileRegs<ROWS>& r,
                                           unsigned short* __restrict__ lds, int tid) {
#pragma unroll
  for (int it = 0; it < (ROWS * 8) / 256; ++it) {
    const int id = tid + it * 256;
    uint2 p;
    p.x = pack_bf16(r.v[it].x, r.v[it].y);
    p.y = pack_bf16(r.v[it].z, r.v[it].w);
    *reinterpret_cast<uint2*>(lds + (id >> 3) * LDSW + ((id & 7) << 2)) = p;
  }
}

// Load one 16x32 bf16 fragment row for this lane (ISA 7.12.2 16-bit A/B layout):
// lane<16: K 0..7 + 16..23 ; lane>=16: K 8..15 + 24..31 (two b128 LDS reads).
__device__ __forceinline__ v16bf load_frag(const unsigned short* rowp, int half) {
  union { uint4 u[2]; v16bf v; } t;
  t.u[0] = *reinterpret_cast<const uint4*>(rowp + half * 8);
  t.u[1] = *reinterpret_cast<const uint4*>(rowp + 16 + half * 8);
  return t.v;
}

// Build a bf16 B-fragment directly from a global row-major [n][k] row (fp32).
__device__ __forceinline__ v16bf bfrag_global(const float* __restrict__ rowb, int half) {
  union { uint4 u[2]; v16bf v; } t;
  const float* c0 = rowb + half * 8;
  const float* c1 = rowb + 16 + half * 8;
  const float4 f0 = *reinterpret_cast<const float4*>(c0);
  const float4 f1 = *reinterpret_cast<const float4*>(c0 + 4);
  const float4 f2 = *reinterpret_cast<const float4*>(c1);
  const float4 f3 = *reinterpret_cast<const float4*>(c1 + 4);
  t.u[0].x = pack_bf16(f0.x, f0.y); t.u[0].y = pack_bf16(f0.z, f0.w);
  t.u[0].z = pack_bf16(f1.x, f1.y); t.u[0].w = pack_bf16(f1.z, f1.w);
  t.u[1].x = pack_bf16(f2.x, f2.y); t.u[1].y = pack_bf16(f2.z, f2.w);
  t.u[1].z = pack_bf16(f3.x, f3.y); t.u[1].w = pack_bf16(f3.z, f3.w);
  return t.v;
}

template<int MS, int NS>
__device__ __forceinline__ void mma_step(const unsigned short* __restrict__ As,
                                         const unsigned short* __restrict__ Bs,
                                         v8f (&acc)[MS][NS],
                                         int arow0, int brow0, int rl, int half) {
  v16bf af[MS], bfr[NS];
#pragma unroll
  for (int i = 0; i < MS; ++i)
    af[i] = load_frag(As + (arow0 + i * 16 + rl) * LDSW, half);
#pragma unroll
  for (int j = 0; j < NS; ++j)
    bfr[j] = load_frag(Bs + (brow0 + j * 16 + rl) * LDSW, half);
#pragma unroll
  for (int i = 0; i < MS; ++i)
#pragma unroll
    for (int j = 0; j < NS; ++j)
      acc[i][j] = __builtin_amdgcn_wmma_f32_16x16x32_bf16(
          false, af[i], false, bfr[j], (short)0, acc[i][j], false, false);
}

// Generic TN GEMM (projections + output projection).
// STORE_MODE 0: C[m*ldc + n] = (acc + bias[n]) * scale
// STORE_MODE 1: v-transpose store -> vT[b][h][dk][kv]
// K must be a multiple of 64.
template<int BM, int BN, int WROWS, int WCOLS, int BIAS, int STORE_MODE>
__global__ __launch_bounds__(256) void gemm_tn(
    const float* __restrict__ A, const float* __restrict__ Bm,
    const float* __restrict__ bias, float* __restrict__ C,
    long long lda, long long ldb, long long ldc, int K, float scale)
{
  constexpr int MS = BM / (16 * WROWS);
  constexpr int NS = BN / (16 * WCOLS);

  __shared__ __align__(16) unsigned short As[2 * BM * LDSW];
  __shared__ __align__(16) unsigned short Bs[2 * BN * LDSW];

  const int tid  = threadIdx.x;
  const int wave = tid >> 5;
  const int lane = tid & 31;
  const int wm   = wave / WCOLS;
  const int wn   = wave % WCOLS;
  const int half = lane >> 4;
  const int rl   = lane & 15;
  const int arow0 = wm * MS * 16;
  const int brow0 = wn * NS * 16;

  const float* Ag = A  + (long long)blockIdx.y * BM * lda;
  const float* Bg = Bm + (long long)blockIdx.x * BN * ldb;

  v8f acc[MS][NS] = {};

  TileRegs<BM> rA0, rA1;
  TileRegs<BN> rB0, rB1;
  tile_load<BM>(rA0, Ag, lda, tid);
  tile_load<BN>(rB0, Bg, ldb, tid);

  for (int kt = 0; kt < K; kt += 64) {
    tile_store<BM>(rA0, As, tid);
    tile_store<BN>(rB0, Bs, tid);
    if (kt + 32 < K) {
      tile_load<BM>(rA1, Ag + kt + 32, lda, tid);
      tile_load<BN>(rB1, Bg + kt + 32, ldb, tid);
    }
    if (kt + 96 < K)  // prefetch two tiles ahead (global_prefetch_b8)
      __builtin_prefetch(Ag + (long long)(tid & (BM - 1)) * lda + kt + 96, 0, 1);
    __syncthreads();
    mma_step<MS, NS>(As, Bs, acc, arow0, brow0, rl, half);

    tile_store<BM>(rA1, As + BM * LDSW, tid);
    tile_store<BN>(rB1, Bs + BN * LDSW, tid);
    if (kt + 64 < K) {
      tile_load<BM>(rA0, Ag + kt + 64, lda, tid);
      tile_load<BN>(rB0, Bg + kt + 64, ldb, tid);
    }
    __syncthreads();
    mma_step<MS, NS>(As + BM * LDSW, Bs + BN * LDSW, acc, arow0, brow0, rl, half);
  }

  // Epilogue. C frag layout: elem e -> M = e + 8*half, N = lane%16.
#pragma unroll
  for (int i = 0; i < MS; ++i)
#pragma unroll
    for (int j = 0; j < NS; ++j)
#pragma unroll
      for (int e = 0; e < 8; ++e) {
        long long m = (long long)blockIdx.y * BM + wm * MS * 16 + i * 16 + half * 8 + e;
        long long n = (long long)blockIdx.x * BN + wn * NS * 16 + j * 16 + rl;
        float v = acc[i][j][e];
        if (BIAS) v += bias[n];
        v *= scale;
        if (STORE_MODE == 0) {
          C[m * ldc + n] = v;
        } else {               // vT[b][h][dk][kv], M = b*NKV+kv, N = h*DK+dk
          int b2 = (int)(m >> 11), kv = (int)(m & 2047);
          int hh = (int)(n >> 6),  dk = (int)(n & 63);
          C[(((long long)b2 * H_ + hh) * DK + dk) * NKV_ + kv] = v;
        }
      }
}

// ---------------------------------------------------------------------------
// Fused attention core: one block = 16 q-rows x all 2048 kv for one (b,h).
//   scores = q @ k^T  (WMMA, k rows from L2)  -> mask -> softmax
//   attn written to HBM once (fp32); probs also kept as bf16 in LDS
//   ctx = probs @ vT  (WMMA, vT rows from L2), deterministic wave reduction.
// Dynamic LDS layout (bytes):
//   [0      ) qf     16x64 fp32 q tile        4096
//   [4096   ) As     16x72 bf16 q tile        2304 (+pad 6400)
//   [6400   ) redbuf 8x16 fp32                 512
//   [6912   ) grow   16 fp32                    64
//   [6976   ) probs  8 waves x 16 x 264 bf16  67584
//   [74560  ) ctxbuf 8 waves x 16 x 64 fp32   32768   => total 107328
// ---------------------------------------------------------------------------
#define SM_AS    4096
#define SM_RED   6400
#define SM_GROW  6912
#define SM_PROBS 6976
#define SM_CTX   74560
#define SM_TOTAL 107328
#define PW_      264   // probs row stride (ushorts): 528 B = 33*16 B

__global__ __launch_bounds__(256) void attn_core(
    const float* __restrict__ q, const float* __restrict__ k,
    const float* __restrict__ vT, const unsigned char* __restrict__ mask,
    float* __restrict__ attn, float* __restrict__ ctx)
{
  extern __shared__ char smem[];
  float*          qf     = (float*)smem;
  unsigned short* As     = (unsigned short*)(smem + SM_AS);
  float*          redbuf = (float*)(smem + SM_RED);     // [wave][16]
  float*          grow   = (float*)(smem + SM_GROW);
  unsigned short* probs  = (unsigned short*)(smem + SM_PROBS);
  float*          ctxbuf = (float*)(smem + SM_CTX);     // [wave][16*64]

  const int m0 = blockIdx.x * 16;
  const int bz = blockIdx.y;                 // b*H + h
  const int b  = bz >> 4;
  const int h  = bz & 15;

  const float* qp  = q  + (long long)b * NQ_ * DM + (long long)m0 * DM + h * DK;
  const float* kp  = k  + (long long)b * NKV_ * DM + h * DK;
  const float* vTp = vT + (long long)bz * DK * NKV_;
  const unsigned char* mk = mask + (long long)b * NKV_;
  float* ap   = attn + ((long long)bz * NQ_ + m0) * NKV_;
  float* ctxp = ctx + ((long long)b * NQ_ + m0) * DM + h * DK;

  const int tid  = threadIdx.x;
  const int wave = tid >> 5;
  const int lane = tid & 31;
  const int half = lane >> 4;
  const int rl   = lane & 15;
  const int strip = wave * 256;              // this wave's kv range
  const float NEG = -__builtin_inff();

  // Stage 16x64 fp32 q tile into LDS with async-to-LDS (ASYNCcnt), 1 b128/thread.
  {
    const unsigned laddr = (unsigned)(size_t)qf + (unsigned)(tid << 4);
    const float* ga = qp + (long long)(tid >> 4) * DM + ((tid & 15) << 2);
    asm volatile("global_load_async_to_lds_b128 %0, %1, off"
                 :: "v"(laddr), "v"(ga) : "memory");
    asm volatile("s_wait_asynccnt 0x0" ::: "memory");
  }
  __syncthreads();
  {
    const int row = tid >> 4, c4 = (tid & 15) << 2;
    const float* s = qf + row * DK + c4;
    uint2 p;
    p.x = pack_bf16(s[0], s[1]);
    p.y = pack_bf16(s[2], s[3]);
    *reinterpret_cast<uint2*>(As + row * 72 + c4) = p;
  }
  __syncthreads();

  v16bf af0 = load_frag(As + rl * 72, half);        // K 0..31
  v16bf af1 = load_frag(As + rl * 72 + 32, half);   // K 32..63

  // ---- scores = q @ k^T
  v8f acc[16] = {};
#pragma unroll
  for (int j = 0; j < 16; ++j) {
    const float* rowb = kp + (long long)(strip + j * 16 + rl) * DM;
    acc[j] = __builtin_amdgcn_wmma_f32_16x16x32_bf16(
        false, af0, false, bfrag_global(rowb, half), (short)0, acc[j], false, false);
    acc[j] = __builtin_amdgcn_wmma_f32_16x16x32_bf16(
        false, af1, false, bfrag_global(rowb + 32, half), (short)0, acc[j], false, false);
  }

  // ---- mask + per-row max. Frag: elem e -> row half*8+e, col strip+j*16+rl.
  float pm[8];
#pragma unroll
  for (int e = 0; e < 8; ++e) pm[e] = NEG;
#pragma unroll
  for (int j = 0; j < 16; ++j) {
    const bool ok = mk[strip + j * 16 + rl] != 0;
#pragma unroll
    for (int e = 0; e < 8; ++e) {
      const float s = ok ? acc[j][e] : NEG;
      acc[j][e] = s;
      pm[e] = fmaxf(pm[e], s);
    }
  }
#pragma unroll
  for (int off = 1; off < 16; off <<= 1)
#pragma unroll
    for (int e = 0; e < 8; ++e) pm[e] = fmaxf(pm[e], __shfl_xor(pm[e], off));
  if (rl == 0)
#pragma unroll
    for (int e = 0; e < 8; ++e) redbuf[wave * 16 + half * 8 + e] = pm[e];
  __syncthreads();
  if (tid < 16) {
    float m = redbuf[tid];
#pragma unroll
    for (int w = 1; w < 8; ++w) m = fmaxf(m, redbuf[w * 16 + tid]);
    grow[tid] = m;
  }
  __syncthreads();
  float rmax[8];
#pragma unroll
  for (int e = 0; e < 8; ++e) rmax[e] = grow[half * 8 + e];

  // ---- exp + per-row sum (masked: -inf -> 0)
  float ps[8];
#pragma unroll
  for (int e = 0; e < 8; ++e) ps[e] = 0.f;
#pragma unroll
  for (int j = 0; j < 16; ++j)
#pragma unroll
    for (int e = 0; e < 8; ++e) {
      const float ev = __expf(acc[j][e] - rmax[e]);
      acc[j][e] = ev;
      ps[e] += ev;
    }
#pragma unroll
  for (int off = 1; off < 16; off <<= 1)
#pragma unroll
    for (int e = 0; e < 8; ++e) ps[e] += __shfl_xor(ps[e], off);
  if (rl == 0)
#pragma unroll
    for (int e = 0; e < 8; ++e) redbuf[wave * 16 + half * 8 + e] = ps[e];
  __syncthreads();
  if (tid < 16) {
    float s = redbuf[tid];
#pragma unroll
    for (int w = 1; w < 8; ++w) s += redbuf[w * 16 + tid];
    grow[tid] = s;
  }
  __syncthreads();
  float inv[8];
#pragma unroll
  for (int e = 0; e < 8; ++e) inv[e] = 1.f / grow[half * 8 + e];

  // ---- write attn (fp32, the only HBM pass) + bf16 probs into this wave's LDS
  unsigned short* pw = probs + wave * (16 * PW_);
#pragma unroll
  for (int j = 0; j < 16; ++j) {
    const int colL = j * 16 + rl;
#pragma unroll
    for (int e = 0; e < 8; ++e) {
      const float pv = acc[j][e] * inv[e];
      ap[(long long)(half * 8 + e) * NKV_ + (strip + colL)] = pv;
      pw[(half * 8 + e) * PW_ + colL] = bf16_one(pv);
    }
  }
  // pw region is wave-private; DS ops are in-order per wave -> no barrier.

  // ---- ctx partial: probs(16x256) @ vT(64x256)^T, 8 K-steps x 4 N-subtiles
  v8f acc2[4] = {};
#pragma unroll
  for (int ks = 0; ks < 8; ++ks) {
    const v16bf pa = load_frag(pw + rl * PW_ + ks * 32, half);
#pragma unroll
    for (int j2 = 0; j2 < 4; ++j2) {
      const float* rowv = vTp + (long long)(j2 * 16 + rl) * NKV_ + strip + ks * 32;
      acc2[j2] = __builtin_amdgcn_wmma_f32_16x16x32_bf16(
          false, pa, false, bfrag_global(rowv, half), (short)0, acc2[j2], false, false);
    }
  }

  // ---- deterministic cross-wave reduction of the 16x64 ctx tile
  float* cb = ctxbuf + wave * (16 * 64);
#pragma unroll
  for (int j2 = 0; j2 < 4; ++j2)
#pragma unroll
    for (int e = 0; e < 8; ++e)
      cb[(half * 8 + e) * 64 + j2 * 16 + rl] = acc2[j2][e];
  __syncthreads();
  {
    float4 s = *reinterpret_cast<const float4*>(ctxbuf + tid * 4);
#pragma unroll
    for (int w = 1; w < 8; ++w) {
      const float4 t4 = *reinterpret_cast<const float4*>(ctxbuf + w * (16 * 64) + tid * 4);
      s.x += t4.x; s.y += t4.y; s.z += t4.z; s.w += t4.w;
    }
    const int row = tid >> 4, d0 = (tid & 15) << 2;
    *reinterpret_cast<float4*>(ctxp + (long long)row * DM + d0) = s;
  }
}

extern "C" void kernel_launch(void* const* d_in, const int* in_sizes, int n_in,
                              void* d_out, int out_size, void* d_ws, size_t ws_size,
                              hipStream_t stream) {
  (void)in_sizes; (void)n_in; (void)out_size; (void)ws_size;
  const float* Q   = (const float*)d_in[0];
  const float* Kin = (const float*)d_in[1];
  const float* V   = (const float*)d_in[2];
  const unsigned char* mask = (const unsigned char*)d_in[3];   // jax bool (1 B/elt)
  const float* Wq = (const float*)d_in[4];
  const float* bq = (const float*)d_in[5];
  const float* Wk = (const float*)d_in[6];
  const float* bk = (const float*)d_in[7];
  const float* Wv = (const float*)d_in[8];
  const float* bv = (const float*)d_in[9];
  const float* Wo = (const float*)d_in[10];
  const float* bo = (const float*)d_in[11];

  float* out  = (float*)d_out;                                   // (B,NQ,DM)
  float* attn = out + (long long)B_ * NQ_ * DM;                  // (B,H,NQ,NKV)

  float* q   = (float*)d_ws;                                     // (B*NQ, DM), pre-scaled
  float* k   = q   + (long long)B_ * NQ_  * DM;                  // (B*NKV, DM)
  float* vT  = k   + (long long)B_ * NKV_ * DM;                  // (B,H,DK,NKV)
  float* ctx = vT  + (long long)B_ * H_ * DK * NKV_;             // (B*NQ, DM)

  const dim3 blk(256, 1, 1);
  const float rs = 0.125f;                                       // 1/sqrt(DK)

  // 1) q = (Q @ Wq^T + bq) * rs
  gemm_tn<128,128,2,4,1,0><<<dim3(DM/128, (B_*NQ_)/128, 1), blk, 0, stream>>>(
      Q, Wq, bq, q, DM, DM, DM, DM, rs);
  // 2) k = K @ Wk^T + bk
  gemm_tn<128,128,2,4,1,0><<<dim3(DM/128, (B_*NKV_)/128, 1), blk, 0, stream>>>(
      Kin, Wk, bk, k, DM, DM, DM, DM, 1.0f);
  // 3) vT = transpose_per_head(V @ Wv^T + bv)
  gemm_tn<128,128,2,4,1,1><<<dim3(DM/128, (B_*NKV_)/128, 1), blk, 0, stream>>>(
      V, Wv, bv, vT, DM, DM, DM, DM, 1.0f);
  // 4) fused: attn = softmax(mask(q @ k^T)); ctx = attn @ v
  attn_core<<<dim3(NQ_/16, B_*H_, 1), blk, SM_TOTAL, stream>>>(
      q, k, vT, mask, attn, ctx);
  // 5) out = ctx @ Wo^T + bo
  gemm_tn<128,128,2,4,1,0><<<dim3(DM/128, (B_*NQ_)/128, 1), blk, 0, stream>>>(
      ctx, Wo, bo, out, DM, DM, DM, DM, 1.0f);
}
